// SelfAttention_56942676410873
// MI455X (gfx1250) — matrix operand order repaired and checked by reading, not compiled, not run on hardware
//
#include <hip/hip_runtime.h>
#include <hip/hip_bf16.h>

#define D_MODEL 1024
#define N_HEADS 16
#define DK      64
#define SEQ     2048
#define BATCH   4
#define M_TOTAL (BATCH * SEQ)   // 8192

typedef __bf16 bf16_t;
typedef __attribute__((ext_vector_type(16))) bf16_t v16bf;
typedef __attribute__((ext_vector_type(8)))  float  v8f;

union AFragU { unsigned int u[8]; v16bf v; };
union CFragU { float f[8]; v8f v; };

__device__ __forceinline__ unsigned short f2bf(float f) {
    unsigned int u = __float_as_uint(f);
    unsigned int r = u + 0x7FFFu + ((u >> 16) & 1u);   // round-to-nearest-even
    return (unsigned short)(r >> 16);
}

// 16-lane-half XOR-pattern reductions via ds_swizzle (immediate mode, no addr VGPR)
__device__ __forceinline__ float red_max16(float x) {
    x = fmaxf(x, __int_as_float(__builtin_amdgcn_ds_swizzle(__float_as_int(x), 0x041F)));
    x = fmaxf(x, __int_as_float(__builtin_amdgcn_ds_swizzle(__float_as_int(x), 0x081F)));
    x = fmaxf(x, __int_as_float(__builtin_amdgcn_ds_swizzle(__float_as_int(x), 0x101F)));
    x = fmaxf(x, __int_as_float(__builtin_amdgcn_ds_swizzle(__float_as_int(x), 0x201F)));
    return x;
}
__device__ __forceinline__ float red_sum16(float x) {
    x += __int_as_float(__builtin_amdgcn_ds_swizzle(__float_as_int(x), 0x041F));
    x += __int_as_float(__builtin_amdgcn_ds_swizzle(__float_as_int(x), 0x081F));
    x += __int_as_float(__builtin_amdgcn_ds_swizzle(__float_as_int(x), 0x101F));
    x += __int_as_float(__builtin_amdgcn_ds_swizzle(__float_as_int(x), 0x201F));
    return x;
}

// ---------------------------------------------------------------------------
// fp32 -> bf16 pack
// ---------------------------------------------------------------------------
__global__ void pack_bf16_kernel(const float* __restrict__ in,
                                 unsigned short* __restrict__ out, int n) {
    int i = blockIdx.x * blockDim.x + threadIdx.x;
    int stride = gridDim.x * blockDim.x;
    for (; i < n; i += stride) out[i] = f2bf(in[i]);
}

// ---------------------------------------------------------------------------
// GEMM: Y[m,n] = sum_k A[m,k] * W[n,k] + bias[n]
// A: bf16 [M x 1024] row-major (uint-packed), W: bf16 [1024 x 1024] row-major.
// Block tile 128x128, K-step 32, double-buffered LDS (1 barrier / K-step).
// 8 waves, each wave computes 32x64 (2x4 WMMA C-frags).
// MODE 0: store bf16 into [B,H,S,dk] layout (QKV).  MODE 1: store f32 [M,N].
// ---------------------------------------------------------------------------
template <int MODE>
__global__ __launch_bounds__(256)
void gemm_kernel(const unsigned int* __restrict__ A,
                 const unsigned int* __restrict__ W,
                 const float* __restrict__ bias,
                 unsigned short* __restrict__ out_bf,
                 float* __restrict__ out_f32) {
    __shared__ unsigned int As[2][128 * 20];  // 32 bf16 (16 uints) + pad4 per row
    __shared__ unsigned int Bs[2][128 * 20];

    const int tid   = threadIdx.x;
    const int lane  = tid & 31;
    const int wave  = tid >> 5;
    const int wm    = wave >> 1;            // 0..3
    const int wn    = wave & 1;             // 0..1
    const int m0    = blockIdx.x * 128;
    const int n0    = blockIdx.y * 128;
    const int lrow  = lane & 15;
    const int lhalf = lane >> 4;

    CFragU acc[2][4];
#pragma unroll
    for (int i = 0; i < 2; i++)
#pragma unroll
        for (int j = 0; j < 4; j++)
#pragma unroll
            for (int r = 0; r < 8; r++) acc[i][j].f[r] = 0.f;

    const int ldr = tid >> 1;               // 0..127 tile row
    const int ldc = (tid & 1) * 8;          // uint col 0 or 8
    const unsigned int* ga = A + (size_t)(m0 + ldr) * 512 + ldc;
    const unsigned int* gb = W + (size_t)(n0 + ldr) * 512 + ldc;

    // prolog: tile 0 into buffer 0
    uint4 a0 = *(const uint4*)(ga);
    uint4 a1 = *(const uint4*)(ga + 4);
    uint4 b0 = *(const uint4*)(gb);
    uint4 b1 = *(const uint4*)(gb + 4);
    *(uint4*)&As[0][ldr * 20 + ldc]     = a0;
    *(uint4*)&As[0][ldr * 20 + ldc + 4] = a1;
    *(uint4*)&Bs[0][ldr * 20 + ldc]     = b0;
    *(uint4*)&Bs[0][ldr * 20 + ldc + 4] = b1;

    for (int it = 0; it < 32; ++it) {
        const int p = it & 1;
        if (it < 31) {                      // prefetch next tile into registers
            const unsigned int* gan = ga + ((it + 1) << 4);
            const unsigned int* gbn = gb + ((it + 1) << 4);
            a0 = *(const uint4*)(gan);
            a1 = *(const uint4*)(gan + 4);
            b0 = *(const uint4*)(gbn);
            b1 = *(const uint4*)(gbn + 4);
        }
        __syncthreads();                    // tile 'it' stores visible; prev reads done

        AFragU af[2], bfr[4];
#pragma unroll
        for (int i = 0; i < 2; i++) {
            int row = wm * 32 + i * 16 + lrow;
#pragma unroll
            for (int v = 0; v < 8; v++)
                af[i].u[v] = As[p][row * 20 + ((v >> 2) * 8 + lhalf * 4 + (v & 3))];
        }
#pragma unroll
        for (int j = 0; j < 4; j++) {
            int row = wn * 64 + j * 16 + lrow;
#pragma unroll
            for (int v = 0; v < 8; v++)
                bfr[j].u[v] = Bs[p][row * 20 + ((v >> 2) * 8 + lhalf * 4 + (v & 3))];
        }
#pragma unroll
        for (int i = 0; i < 2; i++)
#pragma unroll
            for (int j = 0; j < 4; j++)
                acc[i][j].v = __builtin_amdgcn_wmma_f32_16x16x32_bf16(
                    false, af[i].v, false, bfr[j].v, (short)0, acc[i][j].v,
                    false, false);

        if (it < 31) {                      // stage next tile into other buffer
            const int q = p ^ 1;
            *(uint4*)&As[q][ldr * 20 + ldc]     = a0;
            *(uint4*)&As[q][ldr * 20 + ldc + 4] = a1;
            *(uint4*)&Bs[q][ldr * 20 + ldc]     = b0;
            *(uint4*)&Bs[q][ldr * 20 + ldc + 4] = b1;
        }
    }

#pragma unroll
    for (int i = 0; i < 2; i++) {
#pragma unroll
        for (int j = 0; j < 4; j++) {
            int nn = n0 + wn * 64 + j * 16 + lrow;
            float bv = bias[nn];
#pragma unroll
            for (int r = 0; r < 8; r++) {
                int mm = m0 + wm * 32 + i * 16 + r + lhalf * 8;
                float y = acc[i][j].f[r] + bv;
                if (MODE == 0) {
                    int b = mm >> 11, s = mm & (SEQ - 1);
                    int h = nn >> 6,  d = nn & (DK - 1);
                    out_bf[(((size_t)(b * N_HEADS + h) * SEQ + s) << 6) + d] =
                        f2bf(y);
                } else {
                    out_f32[(size_t)mm * 1024 + nn] = y;
                }
            }
        }
    }
}

// ---------------------------------------------------------------------------
// Flash attention (causal). Q/K/V: bf16 [B,H,S,64]. O: bf16 [B,S,1024].
// Block: 256 threads = 8 waves, one q-tile of 128 rows of one (b,h).
// Each wave owns 16 query rows; iterate over 32-key blocks (skip above diag).
// Softmax in exp2 domain; mask applied only on diagonal-straddling blocks.
// ---------------------------------------------------------------------------
__global__ __launch_bounds__(256)
void flash_attn_kernel(const unsigned int* __restrict__ Q,
                       const unsigned int* __restrict__ K,
                       const unsigned int* __restrict__ V,
                       unsigned short* __restrict__ O) {
    __shared__ unsigned int   Ks[32 * 36];      // [key][dim] 64 bf16 + pad
    __shared__ unsigned short Vt[64 * 36];      // [dim][key] transposed + pad
    __shared__ unsigned short Ps[8 * 16 * 36];  // per-wave P relayout scratch

    const int tid   = threadIdx.x;
    const int lane  = tid & 31;
    const int wave  = tid >> 5;
    const int lrow  = lane & 15;
    const int lhalf = lane >> 4;
    const int qt    = blockIdx.x;
    const int bh    = blockIdx.y;               // b*16 + h
    const int b     = bh >> 4, h = bh & 15;

    const size_t head_base = (size_t)bh * SEQ * 32;   // 32 uints per row
    const int q0 = qt * 128 + wave * 16;

    // Q frags (16 x 64) kept in registers
    AFragU qa[2];
#pragma unroll
    for (int i = 0; i < 2; i++) {
        const unsigned int* qrow = Q + head_base + (size_t)(q0 + lrow) * 32 + i * 16;
#pragma unroll
        for (int v = 0; v < 8; v++)
            qa[i].u[v] = qrow[(v >> 2) * 8 + lhalf * 4 + (v & 3)];
    }

    CFragU acc[4];
#pragma unroll
    for (int j = 0; j < 4; j++)
#pragma unroll
        for (int r = 0; r < 8; r++) acc[j].f[r] = 0.f;
    float mrow[8], lsum[8];
#pragma unroll
    for (int r = 0; r < 8; r++) { mrow[r] = -__builtin_inff(); lsum[r] = 0.f; }

    const int   kend   = qt * 128 + 128;
    const int   qmax   = q0 + 15;
    // softmax in exp2 domain: fold 1/sqrt(64) * log2(e) into the score scale
    const float qscale = 0.125f * 1.4426950408889634f;

    for (int kb = 0; kb < kend; kb += 32) {
        {   // cooperative K/V tile load; V transposed into LDS
            int key = tid >> 3;
            int cu  = (tid & 7) * 4;
            uint4 kv = *(const uint4*)(K + head_base + (size_t)(kb + key) * 32 + cu);
            uint4 vv = *(const uint4*)(V + head_base + (size_t)(kb + key) * 32 + cu);
            __syncthreads();                    // previous iter done with LDS
            *(uint4*)&Ks[key * 36 + cu] = kv;
            unsigned int vvu[4] = {vv.x, vv.y, vv.z, vv.w};
#pragma unroll
            for (int j = 0; j < 4; j++) {
                int d0 = cu * 2 + j * 2;
                Vt[(d0    ) * 36 + key] = (unsigned short)(vvu[j] & 0xFFFFu);
                Vt[(d0 + 1) * 36 + key] = (unsigned short)(vvu[j] >> 16);
            }
        }
        __syncthreads();

        if (kb <= qmax) {
            // scores: two 16x16 frags covering keys kb..kb+31
            CFragU s[2];
#pragma unroll
            for (int j = 0; j < 2; j++) {
                AFragU kf0, kf1;
                int krow = j * 16 + lrow;
#pragma unroll
                for (int v = 0; v < 8; v++) {
                    kf0.u[v] = Ks[krow * 36 + ((v >> 2) * 8 + lhalf * 4 + (v & 3))];        // dims 0..31
                    kf1.u[v] = Ks[krow * 36 + 16 + ((v >> 2) * 8 + lhalf * 4 + (v & 3))];   // dims 32..63
                }
                v8f z = {};
                z = __builtin_amdgcn_wmma_f32_16x16x32_bf16(
                        false, qa[0].v, false, kf0.v, (short)0, z, false, false);
                z = __builtin_amdgcn_wmma_f32_16x16x32_bf16(
                        false, qa[1].v, false, kf1.v, (short)0, z, false, false);
                s[j].v = z;
            }
            // scale (+ causal mask only when the block straddles the diagonal)
            if (kb + 31 <= q0) {                // fully unmasked fast path
#pragma unroll
                for (int j = 0; j < 2; j++)
#pragma unroll
                    for (int r = 0; r < 8; r++) s[j].f[r] *= qscale;
            } else {
#pragma unroll
                for (int j = 0; j < 2; j++) {
                    int kcol = kb + j * 16 + lrow;
#pragma unroll
                    for (int r = 0; r < 8; r++) {
                        int qr = q0 + r + lhalf * 8;
                        float sv = s[j].f[r] * qscale;
                        s[j].f[r] = (kcol <= qr) ? sv : -__builtin_inff();
                    }
                }
            }
            // online softmax (exp2 domain), 16-lane-half reductions
            float mnew[8], resc[8];
#pragma unroll
            for (int r = 0; r < 8; r++) {
                float x = red_max16(fmaxf(s[0].f[r], s[1].f[r]));
                mnew[r] = fmaxf(mrow[r], x);
                resc[r] = __builtin_amdgcn_exp2f(mrow[r] - mnew[r]);
                mrow[r] = mnew[r];
            }
#pragma unroll
            for (int j = 0; j < 2; j++)
#pragma unroll
                for (int r = 0; r < 8; r++)
                    s[j].f[r] = __builtin_amdgcn_exp2f(s[j].f[r] - mnew[r]);
#pragma unroll
            for (int r = 0; r < 8; r++) {
                float x = red_sum16(s[0].f[r] + s[1].f[r]);
                lsum[r] = lsum[r] * resc[r] + x;
            }
            // P (C-layout) -> LDS -> A-frag (wave-private; LDS is in-order)
            unsigned short* pw = &Ps[wave * 16 * 36];
#pragma unroll
            for (int j = 0; j < 2; j++)
#pragma unroll
                for (int r = 0; r < 8; r++)
                    pw[(r + lhalf * 8) * 36 + j * 16 + lrow] = f2bf(s[j].f[r]);
            AFragU pf;
            const unsigned int* pr = (const unsigned int*)pw;
#pragma unroll
            for (int v = 0; v < 8; v++)
                pf.u[v] = pr[lrow * 18 + ((v >> 2) * 8 + lhalf * 4 + (v & 3))];
            // P @ V accumulate over dk in 4 N-tiles
            const unsigned int* vt32 = (const unsigned int*)Vt;
#pragma unroll
            for (int j = 0; j < 4; j++) {
                AFragU vf;
                int drow = j * 16 + lrow;
#pragma unroll
                for (int v = 0; v < 8; v++)
                    vf.u[v] = vt32[drow * 18 + ((v >> 2) * 8 + lhalf * 4 + (v & 3))];
#pragma unroll
                for (int r = 0; r < 8; r++) acc[j].f[r] *= resc[r];
                acc[j].v = __builtin_amdgcn_wmma_f32_16x16x32_bf16(
                    false, pf.v, false, vf.v, (short)0, acc[j].v, false, false);
            }
        }
        __syncthreads();
    }

    float inv[8];
#pragma unroll
    for (int r = 0; r < 8; r++) inv[r] = 1.0f / lsum[r];
#pragma unroll
    for (int j = 0; j < 4; j++)
#pragma unroll
        for (int r = 0; r < 8; r++) {
            int qr = q0 + r + lhalf * 8;
            int d  = j * 16 + lrow;
            O[(size_t)(b * SEQ + qr) * 1024 + h * DK + d] =
                f2bf(acc[j].f[r] * inv[r]);
        }
}

// ---------------------------------------------------------------------------
extern "C" void kernel_launch(void* const* d_in, const int* in_sizes, int n_in,
                              void* d_out, int out_size, void* d_ws, size_t ws_size,
                              hipStream_t stream) {
    (void)in_sizes; (void)n_in; (void)out_size; (void)ws_size;
    const float* X  = (const float*)d_in[0];
    const float* Wq = (const float*)d_in[1]; const float* bq = (const float*)d_in[2];
    const float* Wk = (const float*)d_in[3]; const float* bk = (const float*)d_in[4];
    const float* Wv = (const float*)d_in[5]; const float* bv = (const float*)d_in[6];
    const float* Wo = (const float*)d_in[7]; const float* bo = (const float*)d_in[8];

    char* ws = (char*)d_ws;
    size_t off = 0;
    unsigned short* Xbf = (unsigned short*)(ws + off); off += (size_t)M_TOTAL * 1024 * 2;
    unsigned short* Wqb = (unsigned short*)(ws + off); off += (size_t)1024 * 1024 * 2;
    unsigned short* Wkb = (unsigned short*)(ws + off); off += (size_t)1024 * 1024 * 2;
    unsigned short* Wvb = (unsigned short*)(ws + off); off += (size_t)1024 * 1024 * 2;
    unsigned short* Wob = (unsigned short*)(ws + off); off += (size_t)1024 * 1024 * 2;
    unsigned short* Qb  = (unsigned short*)(ws + off); off += (size_t)M_TOTAL * 1024 * 2;
    unsigned short* Kb  = (unsigned short*)(ws + off); off += (size_t)M_TOTAL * 1024 * 2;
    unsigned short* Vb  = (unsigned short*)(ws + off); off += (size_t)M_TOTAL * 1024 * 2;
    unsigned short* Ob  = (unsigned short*)(ws + off); off += (size_t)M_TOTAL * 1024 * 2;

    pack_bf16_kernel<<<4096, 256, 0, stream>>>(X,  Xbf, M_TOTAL * 1024);
    pack_bf16_kernel<<<1024, 256, 0, stream>>>(Wq, Wqb, 1024 * 1024);
    pack_bf16_kernel<<<1024, 256, 0, stream>>>(Wk, Wkb, 1024 * 1024);
    pack_bf16_kernel<<<1024, 256, 0, stream>>>(Wv, Wvb, 1024 * 1024);
    pack_bf16_kernel<<<1024, 256, 0, stream>>>(Wo, Wob, 1024 * 1024);

    dim3 gg(M_TOTAL / 128, 1024 / 128);
    gemm_kernel<0><<<gg, 256, 0, stream>>>((const unsigned int*)Xbf,
                                           (const unsigned int*)Wqb, bq, Qb, nullptr);
    gemm_kernel<0><<<gg, 256, 0, stream>>>((const unsigned int*)Xbf,
                                           (const unsigned int*)Wkb, bk, Kb, nullptr);
    gemm_kernel<0><<<gg, 256, 0, stream>>>((const unsigned int*)Xbf,
                                           (const unsigned int*)Wvb, bv, Vb, nullptr);

    flash_attn_kernel<<<dim3(SEQ / 128, BATCH * N_HEADS), 256, 0, stream>>>(
        (const unsigned int*)Qb, (const unsigned int*)Kb,
        (const unsigned int*)Vb, Ob);

    gemm_kernel<1><<<gg, 256, 0, stream>>>((const unsigned int*)Ob,
                                           (const unsigned int*)Wob, bo,
                                           nullptr, (float*)d_out);
}
